// incidence_matrix_learn_35940286333313
// MI455X (gfx1250) — matrix operand order repaired and checked by reading, not compiled, not run on hardware
//
#include <hip/hip_runtime.h>

// CDNA5 / gfx1250 implementation of incidence_matrix_learn (two scales).
//
// Key simplification (exact): softmax rows sum to 1 => at most one entry > 0.5,
// and it is the row max (always in top-k). So adj_bin has <= 1 nonzero per row,
// P[n,m] reduces to equality of per-node hyperedge labels, and the A*A^T step
// vanishes. The heavy compute is the batched Gram matrix sim = nrm nrm^T / tau
// (~36 GFLOP/call), done with v_wmma_f32_16x16x32_f16 (f16 safe: unit rows),
// fused with the exp-sum / positive-sum epilogue so B*N*N is never stored.

typedef __attribute__((ext_vector_type(16))) _Float16 v16h;
typedef __attribute__((ext_vector_type(8)))  _Float16 v8h;
typedef __attribute__((ext_vector_type(8)))  float    v8f;

#define K_ALPHA   3.0f
#define K_INV_TAU 10.0f

// ---------------- accumulator init (ws is poisoned; must re-zero every call) ----
__global__ void init_acc_kernel(float* lsum, int* valid) {
  if (threadIdx.x == 0) { lsum[0] = 0.f; lsum[1] = 0.f; valid[0] = 0; valid[1] = 0; }
}

// ---------------- adjacency: one block (128 thr) per node ----------------------
// z_h = relu(3 * <enod[n], ehy[h]>) ; p = softmax_h(z) ; adj = (p > 0.5)
// hidx[n] = h with p_h > 0.5, else -1  (at most one such h exists)
__global__ void adj_kernel(const float* __restrict__ enod, const float* __restrict__ ehy,
                           int H, int D, float* __restrict__ adj_out, int* __restrict__ hidx) {
  __shared__ float sv[512];
  __shared__ float red[128];
  const int n = blockIdx.x;
  const int t = threadIdx.x;
  const float* row = enod + (size_t)n * D;
  for (int d = t; d < D; d += 128) sv[d] = row[d];
  if (t == 0) hidx[n] = -1;
  __syncthreads();

  float z = -1e30f;
  if (t < H) {
    const float* hrow = ehy + (size_t)t * D;
    float acc = 0.f;
    for (int d = 0; d < D; ++d) acc = fmaf(sv[d], hrow[d], acc);
    z = fmaxf(K_ALPHA * acc, 0.f);
  }
  red[t] = z; __syncthreads();
  for (int s = 64; s > 0; s >>= 1) { if (t < s) red[t] = fmaxf(red[t], red[t + s]); __syncthreads(); }
  const float zmax = red[0]; __syncthreads();

  const float e = (t < H) ? expf(z - zmax) : 0.f;
  red[t] = e; __syncthreads();
  for (int s = 64; s > 0; s >>= 1) { if (t < s) red[t] += red[t + s]; __syncthreads(); }
  const float sum = red[0];

  if (t < H) {
    const float p = e / sum;
    adj_out[(size_t)n * H + t] = (p > 0.5f) ? 1.0f : 0.0f;
    if (p > 0.5f) hidx[n] = t;   // at most one thread takes this branch
  }
}

// ---------------- positives-per-node count -------------------------------------
__global__ void cnt_kernel(const int* __restrict__ hidx, int N,
                           int* __restrict__ cnt, int* __restrict__ valid) {
  const int n = blockIdx.x * blockDim.x + threadIdx.x;
  if (n >= N) return;
  const int h = hidx[n];
  int c = 0;
  if (h >= 0) {
    for (int m = 0; m < N; ++m) c += (hidx[m] == h) ? 1 : 0;
    c -= 1;  // exclude self
  }
  cnt[n] = c;
  if (c > 0) atomicAdd(valid, 1);
}

// ---------------- normalize rows -> f16 [B,N,D] --------------------------------
__global__ void norm_kernel(const float* __restrict__ x, const float* __restrict__ w,
                            int N, int L, int D, _Float16* __restrict__ out) {
  __shared__ float red[128];
  const int bn = blockIdx.x;            // b*N + n
  const int b = bn / N, n = bn - b * N;
  const float* src = (n < L) ? (x + ((size_t)b * L + n) * D)
                             : (w + (size_t)(n - L) * D);
  const int t = threadIdx.x;
  float ss = 0.f;
  for (int d = t; d < D; d += 128) { const float v = src[d]; ss = fmaf(v, v, ss); }
  red[t] = ss; __syncthreads();
  for (int s = 64; s > 0; s >>= 1) { if (t < s) red[t] += red[t + s]; __syncthreads(); }
  const float nrm = fmaxf(sqrtf(red[0]), 1e-12f);
  const float sc = 1.0f / nrm;
  _Float16* dst = out + (size_t)bn * D;
  for (int d = t; d < D; d += 128) dst[d] = (_Float16)(src[d] * sc);
}

// ---------------- fused Gram + contrastive-loss reduction ----------------------
// One wave per (batch b, 16-row tile I). For each 16-col tile J: 16x16x512 GEMM
// via 16 chained v_wmma_f32_16x16x32_f16, then per-lane exp-sum / positive-sum.
// exp is overflow-safe without max-shift: sim in [-10,10] (unit rows / tau).
__global__ void __launch_bounds__(32)
sim_loss_kernel(const _Float16* __restrict__ nrm, const int* __restrict__ hidx,
                const int* __restrict__ cnt, int N, int D, float* __restrict__ lsum) {
  const int I    = blockIdx.x;          // row tile
  const int b    = blockIdx.y;          // batch
  const int lane = threadIdx.x;         // 0..31 (wave32)
  const int q    = lane & 15;
  const int half = lane >> 4;
  const _Float16* V = nrm + (size_t)b * N * D;

  // Hyperedge labels for this lane's 8 output rows (C layout: row = half*8 + r)
  const int rowbase = I * 16 + half * 8;
  int hm[8];
  #pragma unroll
  for (int r = 0; r < 8; ++r) hm[r] = hidx[rowbase + r];

  float sumexp[8], possum[8];
  #pragma unroll
  for (int r = 0; r < 8; ++r) { sumexp[r] = 0.f; possum[r] = 0.f; }

  const _Float16* arow = V + (size_t)(I * 16 + q) * D;   // A: lane = row M
  const int nTiles = N >> 4;
  for (int J = 0; J < nTiles; ++J) {
    const _Float16* brow = V + (size_t)(J * 16 + q) * D; // B: lane = col N
    v8f c = {};
    #pragma unroll 4
    for (int kb = 0; kb < D; kb += 32) {
      // A 16x32 f16: lanes 0-15 K=0..15, lanes 16-31 K=16..31 (contiguous 32B)
      v16h a = *(const v16h*)(arow + kb + half * 16);
      // B 32x16 f16: lanes 0-15 K=0..15, lanes 16-31 K=16..31 (contiguous 32B)
      v16h bb = *(const v16h*)(brow + kb + half * 16);
      c = __builtin_amdgcn_wmma_f32_16x16x32_f16(false, a, false, bb,
                                                 (short)0, c, false, false);
    }
    const int nglob = J * 16 + q;
    const int hn = hidx[nglob];
    #pragma unroll
    for (int r = 0; r < 8; ++r) {
      const float sim = c[r] * K_INV_TAU;
      sumexp[r] += expf(sim);
      const int m = rowbase + r;
      if (hm[r] >= 0 && hn == hm[r] && m != nglob) possum[r] += sim;
    }
  }

  // xor butterfly over the 16 lanes of each half (bits 0-3 stay within half)
  #pragma unroll
  for (int off = 1; off < 16; off <<= 1) {
    #pragma unroll
    for (int r = 0; r < 8; ++r) {
      sumexp[r] += __shfl_xor(sumexp[r], off, 32);
      possum[r] += __shfl_xor(possum[r], off, 32);
    }
  }

  if (q == 0) {
    float contrib = 0.f;
    #pragma unroll
    for (int r = 0; r < 8; ++r) {
      const int m = rowbase + r;
      const int cm = cnt[m];
      if (cm > 0) contrib += logf(sumexp[r]) - possum[r] / (float)cm;
    }
    atomicAdd(lsum, contrib);
  }
}

// ---------------- final scalar ------------------------------------------------
__global__ void finalize_kernel(const float* lsum, const int* valid, int B,
                                float* out_loss) {
  if (threadIdx.x == 0 && blockIdx.x == 0) {
    const float l0 = (valid[0] > 0) ? lsum[0] / (float)(B * valid[0]) : 0.f;
    const float l1 = (valid[1] > 0) ? lsum[1] / (float)(B * valid[1]) : 0.f;
    *out_loss = l0 + l1;
  }
}

extern "C" void kernel_launch(void* const* d_in, const int* in_sizes, int n_in,
                              void* d_out, int out_size, void* d_ws, size_t ws_size,
                              hipStream_t stream) {
  (void)in_sizes; (void)n_in; (void)out_size; (void)ws_size;
  const float* x0      = (const float*)d_in[0];
  const float* x1      = (const float*)d_in[1];
  const float* word0   = (const float*)d_in[2];
  const float* word1   = (const float*)d_in[3];
  const float* embhy0  = (const float*)d_in[4];
  const float* embhy1  = (const float*)d_in[5];
  const float* embnod0 = (const float*)d_in[6];
  const float* embnod1 = (const float*)d_in[7];

  const int B = 64, D = 512;
  const int L0 = 512, L1 = 256, WSZ = 128;
  const int H0 = 128, H1 = 64;
  const int N0 = L0 + WSZ;   // 640
  const int N1 = L1 + WSZ;   // 384

  float* out    = (float*)d_out;
  float* adj0   = out;                                   // 640*128
  float* adj1   = out + (size_t)N0 * H0;                 // 384*64
  float* outls  = out + (size_t)N0 * H0 + (size_t)N1 * H1;

  // Workspace layout (reused across scales; kernels serialize on the stream):
  //   [0,16)      : float lsum[2], then int valid[2]
  //   [256,...)   : hidx (<=640 int)
  //   [4096,...)  : cnt  (<=640 int)
  //   [8192,...)  : nrm f16 [B, Nmax, D]  (64*640*512*2 = 41.94 MB)
  char* ws   = (char*)d_ws;
  float* lsum = (float*)ws;
  int* valid  = (int*)(ws + 8);
  int* hidx   = (int*)(ws + 256);
  int* cnt    = (int*)(ws + 4096);
  _Float16* nrm = (_Float16*)(ws + 8192);

  init_acc_kernel<<<1, 32, 0, stream>>>(lsum, valid);

  // ---- scale 0 ----
  adj_kernel<<<N0, 128, 0, stream>>>(embnod0, embhy0, H0, D, adj0, hidx);
  cnt_kernel<<<(N0 + 127) / 128, 128, 0, stream>>>(hidx, N0, cnt, valid + 0);
  norm_kernel<<<B * N0, 128, 0, stream>>>(x0, word0, N0, L0, D, nrm);
  {
    dim3 g(N0 / 16, B);
    sim_loss_kernel<<<g, 32, 0, stream>>>(nrm, hidx, cnt, N0, D, lsum + 0);
  }

  // ---- scale 1 (reuses hidx/cnt/nrm; in-order stream makes this safe) ----
  adj_kernel<<<N1, 128, 0, stream>>>(embnod1, embhy1, H1, D, adj1, hidx);
  cnt_kernel<<<(N1 + 127) / 128, 128, 0, stream>>>(hidx, N1, cnt, valid + 1);
  norm_kernel<<<B * N1, 128, 0, stream>>>(x1, word1, N1, L1, D, nrm);
  {
    dim3 g(N1 / 16, B);
    sim_loss_kernel<<<g, 32, 0, stream>>>(nrm, hidx, cnt, N1, D, lsum + 1);
  }

  finalize_kernel<<<1, 32, 0, stream>>>(lsum, valid, B, outls);
}
